// MultiHeadAttention_6133213299052
// MI455X (gfx1250) — matrix operand order repaired and checked by reading, not compiled, not run on hardware
//
#include <hip/hip_runtime.h>
#include <hip/hip_bf16.h>

// ---------------------------------------------------------------------------
// MHA with relative position bias, MI455X (gfx1250, wave32, WMMA).
// cast->f16, WMMA GEMMs for Q/K/V (V stored transposed), qe = Q@rel^T (WMMA),
// fused flash-attention (WMMA scores + WMMA P*V, cooperative async-LDS K/V
// staging), WMMA output projection with bias.
// Data movement uses CDNA5 GLOBAL_LOAD_ASYNC_TO_LDS_B128 + s_wait_asynccnt,
// double-buffered in the GEMM so DMA of slab k+1 overlaps WMMAs on slab k.
// ---------------------------------------------------------------------------

typedef _Float16 h16;
typedef __attribute__((ext_vector_type(16))) _Float16 v16h;
typedef __attribute__((ext_vector_type(8)))  _Float16 v8h;
typedef __attribute__((ext_vector_type(8)))  float    v8f;

#define HIDDEN 1024
#define HEADS  16
#define HD     64
#define SEQ    2048
#define BATCH  2
#define MAXREL 32
#define QEP    80           // padded qe row pitch (5 tiles of 16)

#define GM 4096             // BATCH*SEQ
#define GN 1024
#define GK 1024
#define BT 128              // GEMM block tile
#define KT 32               // GEMM K step
#define LDP 40              // LDS row pitch in halves (32 + 8 pad)

// ---- CDNA5 async memory->LDS DMA ------------------------------------------

// one 16-byte lane-transfer: LDS[lds_addr] = MEM[gaddr]
static __device__ __forceinline__ void async_cp16(unsigned int lds_addr,
                                                  const h16* g) {
  asm volatile("global_load_async_to_lds_b128 %0, %1, off"
               :: "v"(lds_addr), "v"((unsigned long long)(uintptr_t)g)
               : "memory");
}
static __device__ __forceinline__ void wait_async0() {
  asm volatile("s_wait_asynccnt 0" ::: "memory");
}
static __device__ __forceinline__ unsigned int lds_off(const void* p) {
  return (unsigned int)(uintptr_t)p;   // low 32 bits of generic ptr = LDS addr
}

// ---- wave32 WMMA helpers ---------------------------------------------------

static __device__ __forceinline__ v8f wmma32(v16h a, v16h b, v8f c) {
  return __builtin_amdgcn_wmma_f32_16x16x32_f16(
      false, a, false, b, (short)0, c, false, false);
}

static __device__ __forceinline__ v16h frag_from(const h16* lo, const h16* hi) {
  v8h a = *(const v8h*)lo;
  v8h b = *(const v8h*)hi;
  v16h r;
#pragma unroll
  for (int i = 0; i < 8; ++i) { r[i] = a[i]; r[8 + i] = b[i]; }
  return r;
}

// A-fragment 16x32 f16: lane = row (lane&15); K chunks half*8 and 16+half*8
static __device__ __forceinline__ v16h load_a(const h16* src, int ld, int lane) {
  const h16* p = src + (lane & 15) * ld + (lane >> 4) * 8;
  return frag_from(p, p + 16);
}

// B-fragment 32x16 f16 with B[k,n] = src[n*ld + k]: lane = col; K chunks half*8 and 16+half*8
static __device__ __forceinline__ v16h load_b(const h16* src, int ld, int lane) {
  const h16* p = src + (lane & 15) * ld + (lane >> 4) * 8;
  return frag_from(p, p + 16);
}

// ---- elementwise casts -----------------------------------------------------

__global__ void k_cvt(const float* __restrict__ s, h16* __restrict__ d, int n) {
  int i = blockIdx.x * blockDim.x + threadIdx.x;
  if (i < n) d[i] = (h16)s[i];
}

__global__ void k_rel(const float* __restrict__ rel, h16* __restrict__ d) {
  int i = blockIdx.x * blockDim.x + threadIdx.x;   // QEP*HD threads
  if (i >= QEP * HD) return;
  int v = i / HD, k = i % HD;
  d[i] = (v < 2 * MAXREL + 1) ? (h16)rel[v * HD + k] : (h16)0.0f;
}

// ---- tiled f16 NT-GEMM: C[M,N] = A[M,K] * W[N,K]^T -------------------------
// mode 0: f16 out, [B,H,S,hd] layout        (Q, K)
// mode 1: f16 out, [B,H,hd,S] layout (V^T)  (V)
// mode 2: f32 out, row-major [M,N], + bias  (final projection)

__global__ __launch_bounds__(256) void k_gemm(const h16* __restrict__ A,
                                              const h16* __restrict__ W,
                                              void* __restrict__ outp,
                                              const float* __restrict__ bias,
                                              int mode) {
  __shared__ alignas(16) h16 As[2][BT * LDP];
  __shared__ alignas(16) h16 Bs[2][BT * LDP];
  const int tid = threadIdx.x, lane = tid & 31, wave = tid >> 5;
  const int wm = wave & 3, wn = wave >> 2;          // 4x2 wave grid
  const int bm = (blockIdx.x % (GM / BT)) * BT;
  const int bn = (blockIdx.x / (GM / BT)) * BT;
  const int srow = tid >> 1, sseg = (tid & 1) * 16; // stage: 16 halves/thread

  const h16* gA = A + (size_t)(bm + srow) * GK + sseg;
  const h16* gB = W + (size_t)(bn + srow) * GK + sseg;

  // async DMA one K-slab (A and B halves) into buffer `buf`
  auto stage = [&](int buf, int k0) {
    unsigned int la = lds_off(&As[buf][srow * LDP + sseg]);
    unsigned int lb = lds_off(&Bs[buf][srow * LDP + sseg]);
    async_cp16(la,      gA + k0);
    async_cp16(la + 16, gA + k0 + 8);
    async_cp16(lb,      gB + k0);
    async_cp16(lb + 16, gB + k0 + 8);
  };

  v8f acc[2][4];
#pragma unroll
  for (int mt = 0; mt < 2; ++mt)
#pragma unroll
    for (int nt = 0; nt < 4; ++nt) { v8f z = {}; acc[mt][nt] = z; }

  stage(0, 0);
  wait_async0();
  __syncthreads();

  int buf = 0;
  for (int k0 = 0; k0 < GK; k0 += KT) {
    if (k0 + KT < GK) {
      stage(buf ^ 1, k0 + KT);          // overlap DMA with WMMAs below
      if (k0 + 2 * KT < GK) {
        __builtin_prefetch(gA + k0 + 2 * KT, 0, 3);
        __builtin_prefetch(gB + k0 + 2 * KT, 0, 3);
      }
    }

    v16h af[2], bf[4];
#pragma unroll
    for (int mt = 0; mt < 2; ++mt)
      af[mt] = load_a(&As[buf][(wm * 32 + mt * 16) * LDP], LDP, lane);
#pragma unroll
    for (int nt = 0; nt < 4; ++nt)
      bf[nt] = load_b(&Bs[buf][(wn * 64 + nt * 16) * LDP], LDP, lane);
#pragma unroll
    for (int mt = 0; mt < 2; ++mt)
#pragma unroll
      for (int nt = 0; nt < 4; ++nt)
        acc[mt][nt] = wmma32(af[mt], bf[nt], acc[mt][nt]);

    wait_async0();                      // next buffer resident
    __syncthreads();
    buf ^= 1;
  }

#pragma unroll
  for (int mt = 0; mt < 2; ++mt) {
#pragma unroll
    for (int nt = 0; nt < 4; ++nt) {
#pragma unroll
      for (int v = 0; v < 8; ++v) {
        int m = bm + wm * 32 + mt * 16 + v + 8 * (lane >> 4);
        int n = bn + wn * 64 + nt * 16 + (lane & 15);
        float val = acc[mt][nt][v];
        if (mode == 2) {
          ((float*)outp)[(size_t)m * GN + n] = val + bias[n];
        } else {
          int b = m / SEQ, s = m % SEQ, h = n / HD, d = n % HD;
          h16* o = (h16*)outp;
          if (mode == 0)
            o[(((size_t)(b * HEADS + h) * SEQ) + s) * HD + d] = (h16)val;
          else
            o[(((size_t)(b * HEADS + h) * HD) + d) * SEQ + s] = (h16)val;
        }
      }
    }
  }
}

// ---- qe = Q @ rel^T : [B*H, SEQ, QEP] f32 ---------------------------------

__global__ __launch_bounds__(128) void k_qe(const h16* __restrict__ Q,
                                            const h16* __restrict__ rel,
                                            float* __restrict__ qe) {
  const int lane = threadIdx.x & 31, wave = threadIdx.x >> 5;
  const int bh = blockIdx.x >> 5;                 // SEQ/64 = 32 q-blocks
  const int q0 = (blockIdx.x & 31) * 64 + wave * 16;
  const h16* qb = Q + (size_t)bh * SEQ * HD + (size_t)q0 * HD;

  v16h a0 = load_a(qb, HD, lane);
  v16h a1 = load_a(qb + 32, HD, lane);
#pragma unroll
  for (int nt = 0; nt < 5; ++nt) {
    v16h b0 = load_b(rel + nt * 16 * HD, HD, lane);
    v16h b1 = load_b(rel + nt * 16 * HD + 32, HD, lane);
    v8f c = {};
    c = wmma32(a0, b0, c);
    c = wmma32(a1, b1, c);
#pragma unroll
    for (int v = 0; v < 8; ++v) {
      int q = q0 + v + 8 * (lane >> 4);
      qe[((size_t)bh * SEQ + q) * QEP + nt * 16 + (lane & 15)] = c[v];
    }
  }
}

// ---- fused flash attention -------------------------------------------------
// Block = 8 waves x 16 query rows (128 q / block). Per 32-key iteration the
// block cooperatively async-DMAs the K tile (32x64) and V^T tile (64x32) into
// LDS once; every wave reads its WMMA B-fragments from LDS.

#define KSP 72   // Ks row pitch in halves (64 + 8)
#define VSP 40   // Vs row pitch in halves (32 + 8)

__global__ __launch_bounds__(256) void k_flash(const h16* __restrict__ Q,
                                               const h16* __restrict__ K,
                                               const h16* __restrict__ Vt,
                                               const float* __restrict__ qe,
                                               h16* __restrict__ O) {
  __shared__ alignas(16) h16 Ks[32 * KSP];        // [key][d]
  __shared__ alignas(16) h16 Vs[64 * VSP];        // [d][key]
  __shared__ alignas(16) h16 plds[8][16 * 40];    // per-wave 16x32 P, padded
  const int tid = threadIdx.x;
  const int lane = tid & 31, wave = tid >> 5;
  const int half = lane >> 4, lc = lane & 15;
  const int bh = blockIdx.x >> 4;                 // SEQ/128 = 16 q-blocks
  const int b = bh / HEADS, h = bh % HEADS;
  const int q0 = (blockIdx.x & 15) * 128 + wave * 16;

  const h16*   Qb  = Q  + (size_t)bh * SEQ * HD;
  const h16*   Kb  = K  + (size_t)bh * SEQ * HD;
  const h16*   Vb  = Vt + (size_t)bh * HD * SEQ;
  const float* qeb = qe + (size_t)bh * SEQ * QEP;

  v16h qf0 = load_a(Qb + (size_t)q0 * HD, HD, lane);       // K-dim 0..31
  v16h qf1 = load_a(Qb + (size_t)q0 * HD + 32, HD, lane);  // K-dim 32..63

  v8f acc[4];
#pragma unroll
  for (int c4 = 0; c4 < 4; ++c4) { v8f z = {}; acc[c4] = z; }
  float mrow[8], lrow[8];
#pragma unroll
  for (int v = 0; v < 8; ++v) { mrow[v] = -1e30f; lrow[v] = 0.0f; }
  const float scale = 0.125f;                     // 1/sqrt(64)
  h16* pl = &plds[wave][0];

  // staging coords (16B per thread per tile)
  const int kr = tid >> 3, kc = (tid & 7) * 8;    // K: 32 rows x 64 cols
  const int vr = tid >> 2, vc = (tid & 3) * 8;    // V: 64 rows x 32 cols
  const unsigned int lks = lds_off(&Ks[kr * KSP + kc]);
  const unsigned int lvs = lds_off(&Vs[vr * VSP + vc]);

  for (int j0 = 0; j0 < SEQ; j0 += 32) {
    __syncthreads();                              // prior reads of Ks/Vs done
    async_cp16(lks, Kb + (size_t)(j0 + kr) * HD + kc);
    async_cp16(lvs, Vb + (size_t)vr * SEQ + j0 + vc);
    wait_async0();
    __syncthreads();                              // tiles resident for all

    // --- scores: two 16x16 tiles, K-dim reduced in two 32-chunks ---
    v8f s[2];
#pragma unroll
    for (int t = 0; t < 2; ++t) {
      v16h b0 = load_b(&Ks[t * 16 * KSP], KSP, lane);
      v16h b1 = load_b(&Ks[t * 16 * KSP + 32], KSP, lane);
      v8f c = {};
      c = wmma32(qf0, b0, c);
      s[t] = wmma32(qf1, b1, c);
    }
    // --- scale + clipped relative-position bias gather ---
#pragma unroll
    for (int t = 0; t < 2; ++t) {
      int j = j0 + t * 16 + lc;
#pragma unroll
      for (int v = 0; v < 8; ++v) {
        int q = q0 + v + 8 * half;
        int dd = q - j;
        dd = dd < -MAXREL ? -MAXREL : (dd > MAXREL ? MAXREL : dd);
        s[t][v] = s[t][v] * scale + qeb[(size_t)q * QEP + dd + MAXREL];
      }
    }
    // --- online softmax: per-row reductions across owning 16 lanes ---
    float mnew[8], alpha[8];
#pragma unroll
    for (int v = 0; v < 8; ++v) {
      float mx = fmaxf(s[0][v], s[1][v]);
#pragma unroll
      for (int off = 8; off >= 1; off >>= 1)
        mx = fmaxf(mx, __shfl_xor(mx, off, 32));
      mnew[v] = fmaxf(mrow[v], mx);
      alpha[v] = __expf(mrow[v] - mnew[v]);
    }
#pragma unroll
    for (int v = 0; v < 8; ++v) {
      float p0 = __expf(s[0][v] - mnew[v]);
      float p1 = __expf(s[1][v] - mnew[v]);
      s[0][v] = p0; s[1][v] = p1;
      float rs = p0 + p1;
#pragma unroll
      for (int off = 8; off >= 1; off >>= 1)
        rs += __shfl_xor(rs, off, 32);
      lrow[v] = lrow[v] * alpha[v] + rs;
      mrow[v] = mnew[v];
    }
    // --- P (C-layout) -> per-wave LDS -> A-fragment re-layout ---
#pragma unroll
    for (int t = 0; t < 2; ++t)
#pragma unroll
      for (int v = 0; v < 8; ++v)
        pl[(v + 8 * half) * 40 + t * 16 + lc] = (h16)s[t][v];
#pragma unroll
    for (int c4 = 0; c4 < 4; ++c4)
#pragma unroll
      for (int v = 0; v < 8; ++v)
        acc[c4][v] *= alpha[v];
    v16h pf = load_a(pl, 40, lane);
    // --- acc += P * V (V tile in LDS, contiguous key dim) ---
#pragma unroll
    for (int c4 = 0; c4 < 4; ++c4) {
      v16h vf = load_b(&Vs[c4 * 16 * VSP], VSP, lane);
      acc[c4] = wmma32(pf, vf, acc[c4]);
    }
  }
  // --- normalize and store to [B, S, H*hd] f16 for the output projection ---
  h16* Ob = O + (size_t)b * SEQ * HIDDEN + h * HD;
#pragma unroll
  for (int v = 0; v < 8; ++v) {
    float inv = 1.0f / lrow[v];
    int q = q0 + v + 8 * half;
#pragma unroll
    for (int c4 = 0; c4 < 4; ++c4)
      Ob[(size_t)q * HIDDEN + c4 * 16 + lc] = (h16)(acc[c4][v] * inv);
  }
}

// ---------------------------------------------------------------------------

extern "C" void kernel_launch(void* const* d_in, const int* in_sizes, int n_in,
                              void* d_out, int out_size, void* d_ws, size_t ws_size,
                              hipStream_t stream) {
  (void)in_sizes; (void)n_in; (void)out_size; (void)ws_size;
  const float* x   = (const float*)d_in[0];
  const float* Wq  = (const float*)d_in[1];
  const float* Wk  = (const float*)d_in[2];
  const float* Wv  = (const float*)d_in[3];
  const float* Wo  = (const float*)d_in[4];
  const float* bo  = (const float*)d_in[5];
  const float* rel = (const float*)d_in[6];

  char* ws = (char*)d_ws;
  size_t off = 0;
  auto alloc = [&](size_t bytes) {
    void* p = ws + off;
    off = (off + bytes + 255) & ~(size_t)255;
    return p;
  };
  h16*   x16   = (h16*)alloc((size_t)GM * GK * 2);
  h16*   wq16  = (h16*)alloc((size_t)GN * GK * 2);
  h16*   wk16  = (h16*)alloc((size_t)GN * GK * 2);
  h16*   wv16  = (h16*)alloc((size_t)GN * GK * 2);
  h16*   wo16  = (h16*)alloc((size_t)GN * GK * 2);
  h16*   rel16 = (h16*)alloc((size_t)QEP * HD * 2);
  h16*   q16   = (h16*)alloc((size_t)GM * HIDDEN * 2);   // [B,H,S,hd]
  h16*   k16   = (h16*)alloc((size_t)GM * HIDDEN * 2);   // [B,H,S,hd]
  h16*   vt16  = (h16*)alloc((size_t)GM * HIDDEN * 2);   // [B,H,hd,S]
  float* qebuf = (float*)alloc((size_t)BATCH * HEADS * SEQ * QEP * 4);
  h16*   at16  = (h16*)alloc((size_t)GM * HIDDEN * 2);   // [B,S,H*hd]

  // 1) casts
  k_cvt<<<(GM * GK + 255) / 256, 256, 0, stream>>>(x, x16, GM * GK);
  k_cvt<<<(GN * GK + 255) / 256, 256, 0, stream>>>(Wq, wq16, GN * GK);
  k_cvt<<<(GN * GK + 255) / 256, 256, 0, stream>>>(Wk, wk16, GN * GK);
  k_cvt<<<(GN * GK + 255) / 256, 256, 0, stream>>>(Wv, wv16, GN * GK);
  k_cvt<<<(GN * GK + 255) / 256, 256, 0, stream>>>(Wo, wo16, GN * GK);
  k_rel<<<(QEP * HD + 255) / 256, 256, 0, stream>>>(rel, rel16);

  // 2) projections (WMMA GEMMs, async double-buffered staging)
  dim3 gg((GM / BT) * (GN / BT));   // 32 * 8 = 256 blocks
  k_gemm<<<gg, 256, 0, stream>>>(x16, wq16, q16,  nullptr, 0);
  k_gemm<<<gg, 256, 0, stream>>>(x16, wk16, k16,  nullptr, 0);
  k_gemm<<<gg, 256, 0, stream>>>(x16, wv16, vt16, nullptr, 1);

  // 3) qe = Q @ rel^T
  k_qe<<<BATCH * HEADS * (SEQ / 64), 128, 0, stream>>>(q16, rel16, qebuf);

  // 4) fused flash attention
  k_flash<<<BATCH * HEADS * (SEQ / 128), 256, 0, stream>>>(q16, k16, vt16,
                                                           qebuf, at16);

  // 5) output projection + bias -> f32 d_out
  k_gemm<<<gg, 256, 0, stream>>>(at16, wo16, d_out, bo, 2);
}
